// TTLinear_61770219651636
// MI455X (gfx1250) — matrix operand order repaired and checked by reading, not compiled, not run on hardware
//
#include <hip/hip_runtime.h>

// TT-linear on gfx1250: materialize W from TT cores (cheap staged chain),
// then one f16 WMMA GEMM  out[4096,4096] = X[4096,1024] * W[1024,4096] + bias.
//
// ws layout (bytes):
//   A2  f32 [8][8][8][8][16]            @ 0          (262144 B)
//   A3  f32 [8][8][4][8][8][8][16]      @ 262144     (8388608 B)
//   Wt  f16 [4096][1024]  (W transposed)@ 8650752    (8388608 B)
//   Xh  f16 [4096][1024]                @ 17039360   (8388608 B)
// total ~24.3 MB

typedef __attribute__((ext_vector_type(16))) _Float16 v16h;
typedef __attribute__((ext_vector_type(4)))  _Float16 v4h;
typedef __attribute__((ext_vector_type(8)))  float    v8f;

// ---------------------------------------------------------------------------
// Stage 1: A2[m1,m2,n1,n2,r2] = sum_r1 c0[m1,n1,r1] * c1[r1,m2,n2,r2]
// ---------------------------------------------------------------------------
__global__ __launch_bounds__(256) void tt_build_a2(const float* __restrict__ c0,
                                                   const float* __restrict__ c1,
                                                   float* __restrict__ A2) {
    const int tid = blockIdx.x * 256 + threadIdx.x;   // 65536 threads
    const int r2 = tid & 15;
    const int n2 = (tid >> 4) & 7;
    const int n1 = (tid >> 7) & 7;
    const int m2 = (tid >> 10) & 7;
    const int m1 = (tid >> 13) & 7;
    const float* c0p = c0 + (m1 * 8 + n1) * 16;       // core0[0][m1][n1][r1]
    float s = 0.f;
#pragma unroll
    for (int r1 = 0; r1 < 16; ++r1)
        s += c0p[r1] * c1[((r1 * 8 + m2) * 8 + n2) * 16 + r2];
    A2[tid] = s;
}

// ---------------------------------------------------------------------------
// Stage 2: A3[m1,m2,m3,n1,n2,n3,r3] = sum_r2 A2[m1,m2,n1,n2,r2]*c2[r2,m3,n3,r3]
// ---------------------------------------------------------------------------
__global__ __launch_bounds__(256) void tt_build_a3(const float* __restrict__ A2,
                                                   const float* __restrict__ c2,
                                                   float* __restrict__ A3) {
    const int tid = blockIdx.x * 256 + threadIdx.x;   // 2097152 threads
    const int r3 = tid & 15;
    const int n3 = (tid >> 4) & 7;
    const int n2 = (tid >> 7) & 7;
    const int n1 = (tid >> 10) & 7;
    const int m3 = (tid >> 13) & 3;
    const int m2 = (tid >> 15) & 7;
    const int m1 = (tid >> 18) & 7;
    const float* a2p = A2 + (((m1 * 8 + m2) * 8 + n1) * 8 + n2) * 16;
    float s = 0.f;
#pragma unroll
    for (int r2 = 0; r2 < 16; ++r2)
        s += a2p[r2] * c2[((r2 * 4 + m3) * 8 + n3) * 16 + r3];
    A3[tid] = s;
}

// ---------------------------------------------------------------------------
// Stage 3: Wt[j][i] (f16, transposed so GEMM B tile is K-contiguous)
//   W[i,j] = sum_r3 A3[m1,m2,m3,n1,n2,n3,r3] * c3[r3,m4,n4]
// ---------------------------------------------------------------------------
__global__ __launch_bounds__(256) void tt_build_wt(const float* __restrict__ A3,
                                                   const float* __restrict__ c3,
                                                   _Float16* __restrict__ Wt) {
    const int tid = blockIdx.x * 256 + threadIdx.x;   // 4194304 threads
    const int i = tid & 1023;                          // input feature (K)
    const int j = tid >> 10;                           // output feature (N)
    const int m4 = i & 3;
    const int m3 = (i >> 2) & 3;
    const int m2 = (i >> 4) & 7;
    const int m1 = (i >> 7) & 7;
    const int n4 = j & 7;
    const int n3 = (j >> 3) & 7;
    const int n2 = (j >> 6) & 7;
    const int n1 = (j >> 9) & 7;
    const float* a3p = A3 + (((((m1 * 8 + m2) * 4 + m3) * 8 + n1) * 8 + n2) * 8 + n3) * 16;
    float s = 0.f;
#pragma unroll
    for (int r3 = 0; r3 < 16; ++r3)
        s += a3p[r3] * c3[(r3 * 4 + m4) * 8 + n4];
    Wt[tid] = (_Float16)s;                             // Wt[j*1024 + i]
}

// ---------------------------------------------------------------------------
// Stage 4: x fp32 -> fp16
// ---------------------------------------------------------------------------
__global__ __launch_bounds__(256) void tt_cvt_x(const float* __restrict__ x,
                                                _Float16* __restrict__ Xh) {
    const int tid = blockIdx.x * 256 + threadIdx.x;   // 1048576 threads, 4 elems each
    const float4 v = ((const float4*)x)[tid];
    v4h h;
    h.x = (_Float16)v.x; h.y = (_Float16)v.y; h.z = (_Float16)v.z; h.w = (_Float16)v.w;
    ((v4h*)Xh)[tid] = h;
}

// ---------------------------------------------------------------------------
// Stage 5: WMMA GEMM  out[4096,4096] = Xh[4096,1024] x W + bias
//   block tile 128x128, K-step 32, 8 waves (4x2), wave tile 32x64
// ---------------------------------------------------------------------------
__global__ __launch_bounds__(256) void tt_gemm(const _Float16* __restrict__ Xh,
                                               const _Float16* __restrict__ Wt,
                                               const float* __restrict__ bias,
                                               float* __restrict__ out) {
    // padded stride 40 halves (80 B) keeps uint4 alignment and spreads banks
    __shared__ __align__(16) _Float16 sA[128 * 40];
    __shared__ __align__(16) _Float16 sB[128 * 40];

    const int tid   = threadIdx.x;
    const int lane  = tid & 31;
    const int wave  = tid >> 5;      // 0..7
    const int waveM = wave >> 1;     // 0..3  -> 32-row strip
    const int waveN = wave & 1;      // 0..1  -> 64-col strip
    const int rowBlock = blockIdx.y * 128;
    const int colBlock = blockIdx.x * 128;

    v8f acc[2][4];
#pragma unroll
    for (int tm = 0; tm < 2; ++tm)
#pragma unroll
        for (int tn = 0; tn < 4; ++tn)
            acc[tm][tn] = (v8f){0.f, 0.f, 0.f, 0.f, 0.f, 0.f, 0.f, 0.f};

    const int lrow   = lane & 15;           // row/col within 16-tile
    const int khalfA = (lane >> 4) * 16;    // A frag: lanes>=16 hold K+16
    const int kbaseB = (lane >> 4) * 16;    // B frag: lanes>=16 hold K+16

    for (int kt = 0; kt < 32; ++kt) {
        const int k0 = kt * 32;
        // Cooperative tile loads: both tiles are 128 rows x 32 halves,
        // K-contiguous in global (Wt is pre-transposed) -> pure uint4 copies.
#pragma unroll
        for (int it = 0; it < 2; ++it) {
            const int c  = tid + it * 256;       // 512 chunks of 8 halves
            const int r  = c >> 2;               // 0..127
            const int co = (c & 3) * 8;          // 0,8,16,24
            *(uint4*)&sA[r * 40 + co] =
                *(const uint4*)&Xh[(size_t)(rowBlock + r) * 1024 + k0 + co];
            *(uint4*)&sB[r * 40 + co] =
                *(const uint4*)&Wt[(size_t)(colBlock + r) * 1024 + k0 + co];
        }
        __syncthreads();

        // Build fragments (two aligned uint4 LDS loads each)
        v16h a[2], b[4];
#pragma unroll
        for (int tm = 0; tm < 2; ++tm) {
            const _Float16* p = &sA[(waveM * 32 + tm * 16 + lrow) * 40];
            ((uint4*)&a[tm])[0] = *(const uint4*)&p[khalfA];        // K 0..7  / 16..23
            ((uint4*)&a[tm])[1] = *(const uint4*)&p[khalfA + 8];    // K 8..15 / 24..31
        }
#pragma unroll
        for (int tn = 0; tn < 4; ++tn) {
            const _Float16* p = &sB[(waveN * 64 + tn * 16 + lrow) * 40];
            ((uint4*)&b[tn])[0] = *(const uint4*)&p[kbaseB];        // K 0..7  / 16..23
            ((uint4*)&b[tn])[1] = *(const uint4*)&p[kbaseB + 8];    // K 8..15 / 24..31
        }

        // 8 WMMAs per wave per K-step: D = A(16x32) x B(32x16) + C
#pragma unroll
        for (int tm = 0; tm < 2; ++tm)
#pragma unroll
            for (int tn = 0; tn < 4; ++tn)
                acc[tm][tn] = __builtin_amdgcn_wmma_f32_16x16x32_f16(
                    /*neg_a=*/false, a[tm], /*neg_b=*/false, b[tn],
                    /*c_mod=*/(short)0, acc[tm][tn],
                    /*reuse_a=*/false, /*reuse_b=*/false);
        __syncthreads();
    }

    // Epilogue: C layout = VGPR v -> M = v + 8*(lane>=16), N = lane&15
    const int rhalf = (lane >> 4) * 8;
#pragma unroll
    for (int tm = 0; tm < 2; ++tm) {
        const int rowBase = rowBlock + waveM * 32 + tm * 16 + rhalf;
#pragma unroll
        for (int tn = 0; tn < 4; ++tn) {
            const int col = colBlock + waveN * 64 + tn * 16 + lrow;
            const float bv = bias[col];
#pragma unroll
            for (int v = 0; v < 8; ++v)
                out[(size_t)(rowBase + v) * 4096 + col] = acc[tm][tn][v] + bv;
        }
    }
}

// ---------------------------------------------------------------------------
extern "C" void kernel_launch(void* const* d_in, const int* in_sizes, int n_in,
                              void* d_out, int out_size, void* d_ws, size_t ws_size,
                              hipStream_t stream) {
    (void)in_sizes; (void)n_in; (void)out_size; (void)ws_size;

    const float* x    = (const float*)d_in[0];   // [4,1024,1024]
    const float* c0   = (const float*)d_in[1];   // [1,8,8,16]
    const float* c1   = (const float*)d_in[2];   // [16,8,8,16]
    const float* c2   = (const float*)d_in[3];   // [16,4,8,16]
    const float* c3   = (const float*)d_in[4];   // [16,4,8,1]
    const float* bias = (const float*)d_in[5];   // [4096]
    float* out = (float*)d_out;                  // [4,1024,4096]

    char* ws = (char*)d_ws;
    float*    A2 = (float*)(ws);                                   // 262144 B
    float*    A3 = (float*)(ws + 262144);                          // 8388608 B
    _Float16* Wt = (_Float16*)(ws + 262144 + 8388608);             // 8388608 B
    _Float16* Xh = (_Float16*)(ws + 262144 + 8388608 + 8388608);   // 8388608 B

    tt_build_a2<<<dim3(65536 / 256), dim3(256), 0, stream>>>(c0, c1, A2);
    tt_build_a3<<<dim3(2097152 / 256), dim3(256), 0, stream>>>(A2, c2, A3);
    tt_build_wt<<<dim3(4194304 / 256), dim3(256), 0, stream>>>(A3, c3, Wt);
    tt_cvt_x  <<<dim3(1048576 / 256), dim3(256), 0, stream>>>(x, Xh);

    dim3 grid(4096 / 128, 4096 / 128);   // (N tiles, T tiles) = 32x32
    tt_gemm<<<grid, dim3(256), 0, stream>>>(Xh, Wt, bias, out);
}
